// MoE_33732673143329
// MI455X (gfx1250) — matrix operand order repaired and checked by reading, not compiled, not run on hardware
//
#include <hip/hip_runtime.h>
#include <hip/hip_bf16.h>
#include <math.h>

// ---------------------------------------------------------------------------
// MoE GEGLU FFN, top-1 routing.  B=4 C=512 D=1024 N=8 DFF=4096.
// MI455X strategy:
//   * gather tokens per expert (avoid 8x dense compute)
//   * weights (402 MB fp32) are single-use -> stream global->regs,
//     convert fp32->bf16 in-register, feed v_wmma_f32_16x16x32_bf16
//   * 32 rows/block: each converted B fragment feeds 2 WMMAs, keeping
//     cvt VALU (~10us) under the 17us HBM weight-stream floor
//   * row tiles are blockIdx.x (fastest) so blocks sharing weight columns
//     are launch-adjacent -> L2 captures the reuse deterministically
//   * activations (<=19 MB bf16) are L2-resident (192 MB L2) -> no LDS
// ---------------------------------------------------------------------------

#define BB   4
#define CC   512
#define DD   1024
#define NE   8
#define DFF  4096
#define F2   (2 * DFF)       // 8192
#define NT   (BB * CC)       // 2048 tokens
#define RPB  32              // rows per GEMM block (2 x 16-row fragments)
#define TPAD 2304            // capacity: 2048 + 8*31 = 2296 -> 72 tiles of 32
#define ROWT (TPAD / RPB)    // 72 row tiles

typedef __attribute__((ext_vector_type(16))) __bf16 v16bf;
typedef __attribute__((ext_vector_type(8)))  float  v8f;

// ---------------------------------------------------------------------------
// Kernel 0: routing.  One workgroup.  Builds per-expert padded segments:
//   poff[9]  : padded exclusive offsets (multiples of 32)
//   perm[s]  : token index occupying sorted slot s (-1 for pad slots)
//   scl[s]   : combine_mask scale for that token
// ---------------------------------------------------------------------------
__global__ __launch_bounds__(1024) void k_route(
    const float* __restrict__ disp, const float* __restrict__ comb,
    int* __restrict__ poff, int* __restrict__ perm, float* __restrict__ scl) {
  __shared__ int scnt[NE], scnt2[NE], spoff[NE + 1];
  const int tid = threadIdx.x;
  if (tid < NE) { scnt[tid] = 0; scnt2[tid] = 0; }
  __syncthreads();

  // each thread owns tokens tid and tid+1024; one-hot argmax
  int e0 = 0, e1 = 0;
  {
    const float* r = disp + (size_t)tid * NE;
    for (int n = 0; n < NE; ++n) if (r[n] > 0.5f) e0 = n;
    atomicAdd(&scnt[e0], 1);
  }
  {
    const float* r = disp + (size_t)(tid + 1024) * NE;
    for (int n = 0; n < NE; ++n) if (r[n] > 0.5f) e1 = n;
    atomicAdd(&scnt[e1], 1);
  }
  __syncthreads();
  if (tid == 0) {
    int acc = 0;
    for (int e = 0; e < NE; ++e) { spoff[e] = acc; acc += (scnt[e] + RPB - 1) & ~(RPB - 1); }
    spoff[NE] = acc;
    for (int e = 0; e <= NE; ++e) poff[e] = spoff[e];
  }
  __syncthreads();
  for (int s = tid; s < TPAD; s += 1024) perm[s] = -1;
  __syncthreads();
  {
    const int slot = spoff[e0] + atomicAdd(&scnt2[e0], 1);
    perm[slot] = tid;
    scl[slot]  = comb[(size_t)tid * NE + e0];
  }
  {
    const int t = tid + 1024;
    const int slot = spoff[e1] + atomicAdd(&scnt2[e1], 1);
    perm[slot] = t;
    scl[slot]  = comb[(size_t)t * NE + e1];
  }
}

// ---------------------------------------------------------------------------
// Kernel 1: gather + fp32->bf16.  Xb[s, 0:1024] = bf16(hidden[perm[s]]).
// 2 rows per 256-thread block, 8 elements per thread (16B stores).
// ---------------------------------------------------------------------------
__global__ __launch_bounds__(256) void k_gather(
    const float* __restrict__ hid, const int* __restrict__ perm,
    __bf16* __restrict__ Xb) {
  const int tid = threadIdx.x;
  const int s   = blockIdx.x * 2 + (tid >> 7);
  const int c0  = (tid & 127) * 8;
  const int tok = perm[s];
  union { __bf16 h[8]; uint4 q; } o;
  if (tok >= 0) {
    const float4* p = (const float4*)(hid + (size_t)tok * DD + c0);
    float4 a = p[0], b = p[1];
    o.h[0] = (__bf16)a.x; o.h[1] = (__bf16)a.y; o.h[2] = (__bf16)a.z; o.h[3] = (__bf16)a.w;
    o.h[4] = (__bf16)b.x; o.h[5] = (__bf16)b.y; o.h[6] = (__bf16)b.z; o.h[7] = (__bf16)b.w;
  } else {
    o.q = make_uint4(0u, 0u, 0u, 0u);
  }
  *(uint4*)(Xb + (size_t)s * DD + c0) = o.q;
}

// ---------------------------------------------------------------------------
// In-register fp32 -> bf16 conversion of 16 weight values (two runs of 8)
// (B-fragment for one lane: column n, K = kb..kb+7, kb+16..kb+23, pairs packed lo/hi).
// ---------------------------------------------------------------------------
__device__ __forceinline__ v16bf cvt16(const float* __restrict__ p) {
  const float4* q = (const float4*)p;
  const float4 f0 = q[0], f1 = q[1], f2 = q[4], f3 = q[5];
  v16bf r;
  r[0]  = (__bf16)f0.x; r[1]  = (__bf16)f0.y; r[2]  = (__bf16)f0.z; r[3]  = (__bf16)f0.w;
  r[4]  = (__bf16)f1.x; r[5]  = (__bf16)f1.y; r[6]  = (__bf16)f1.z; r[7]  = (__bf16)f1.w;
  r[8]  = (__bf16)f2.x; r[9]  = (__bf16)f2.y; r[10] = (__bf16)f2.z; r[11] = (__bf16)f2.w;
  r[12] = (__bf16)f3.x; r[13] = (__bf16)f3.y; r[14] = (__bf16)f3.z; r[15] = (__bf16)f3.w;
  return r;
}

// Branch-free exact-erf GELU (Abramowitz–Stegun 7.1.26, |err| < 1.5e-7):
// avoids libm erff's divergent exec-masked piecewise expansion.
__device__ __forceinline__ float gelu_exact(float g) {
  const float z = fabsf(g) * 0.70710678118654752f;   // |g|/sqrt(2)
  const float t = __builtin_amdgcn_rcpf(1.0f + 0.3275911f * z);
  float p = 1.061405429f;
  p = p * t - 1.453152027f;
  p = p * t + 1.421413741f;
  p = p * t - 0.284496736f;
  p = p * t + 0.254829592f;
  p = p * t;
  const float er = copysignf(1.0f - p * __expf(-z * z), g);
  return 0.5f * g * (1.0f + er);
}

// ---------------------------------------------------------------------------
// Kernel 2: in-projection + bias + exact-erf GEGLU.
// Block = 4 waves, 32 rows; wave w owns 16 output columns j = by*64+w*16+n.
// Per wave: 2 A fragments (rows 0..15, 16..31) x 2 B streams (x, gate)
//   -> 4 WMMAs per converted-B pair per K-step.
// A (16x32 bf16): lane L -> row m=L%16, K runs {base..base+7, base+16..}
// B (32x16 bf16): lane L -> col n=L%16, K = k0 + (L>>4)*8 + {0..7, 16..23}.
// ---------------------------------------------------------------------------
__global__ __launch_bounds__(128) void k_ffn_in(
    const __bf16* __restrict__ Xb, const float* __restrict__ W_in,
    const float* __restrict__ b_in, const int* __restrict__ poff,
    __bf16* __restrict__ Hb) {
  const int row0 = blockIdx.x * RPB;           // row tile = x (fastest) -> L2 reuse
  if (row0 >= poff[NE]) return;                // block-uniform exit
  int e = 0;
  while (poff[e + 1] <= row0) ++e;             // segment -> expert (<=8 iters)

  const int wave = threadIdx.x >> 5;
  const int lane = threadIdx.x & 31;
  const int n    = lane & 15;
  const int kh   = lane >> 4;                  // K-half select
  const int j    = blockIdx.y * 64 + wave * 16 + n;

  const float*  __restrict__ Wx  = W_in + ((size_t)e * F2 + j) * DD;
  const float*  __restrict__ Wg  = W_in + ((size_t)e * F2 + DFF + j) * DD;
  const __bf16* __restrict__ Ar0 = Xb + (size_t)(row0 + n) * DD + kh * 8;
  const __bf16* __restrict__ Ar1 = Ar0 + (size_t)16 * DD;

  v8f ax0 = {}, ax1 = {}, ag0 = {}, ag1 = {};
  for (int k0 = 0; k0 < DD; k0 += 32) {
    union { uint4 q[2]; v16bf v; } a0, a1;
    a0.q[0] = *(const uint4*)(Ar0 + k0);
    a0.q[1] = *(const uint4*)(Ar0 + k0 + 16);
    a1.q[0] = *(const uint4*)(Ar1 + k0);
    a1.q[1] = *(const uint4*)(Ar1 + k0 + 16);
    const int kb = k0 + kh * 8;
    __builtin_prefetch(Wx + kb + 512, 0, 0);   // global_prefetch_b8, stream ahead
    __builtin_prefetch(Wg + kb + 512, 0, 0);
    const v16bf bx = cvt16(Wx + kb);
    const v16bf bg = cvt16(Wg + kb);
    ax0 = __builtin_amdgcn_wmma_f32_16x16x32_bf16(false, a0.v, false, bx,
                                                  (short)0, ax0, false, false);
    ax1 = __builtin_amdgcn_wmma_f32_16x16x32_bf16(false, a1.v, false, bx,
                                                  (short)0, ax1, false, false);
    ag0 = __builtin_amdgcn_wmma_f32_16x16x32_bf16(false, a0.v, false, bg,
                                                  (short)0, ag0, false, false);
    ag1 = __builtin_amdgcn_wmma_f32_16x16x32_bf16(false, a1.v, false, bg,
                                                  (short)0, ag1, false, false);
  }

  const float bxs = b_in[(size_t)e * F2 + j];
  const float bgs = b_in[(size_t)e * F2 + DFF + j];
#pragma unroll
  for (int r = 0; r < 8; ++r) {
    const int m = r + kh * 8;                  // C layout: M = r + 8*(lane>=16)
    const float h0 = (ax0[r] + bxs) * gelu_exact(ag0[r] + bgs);
    const float h1 = (ax1[r] + bxs) * gelu_exact(ag1[r] + bgs);
    Hb[(size_t)(row0 + m) * DFF + j]      = (__bf16)h0;
    Hb[(size_t)(row0 + 16 + m) * DFF + j] = (__bf16)h1;
  }
}

// ---------------------------------------------------------------------------
// Kernel 3: out-projection + bias + combine-scale + scatter to d_out.
// Same scheme; K = 4096, columns d = by*64 + w*16 + n, 32 rows/block.
// ---------------------------------------------------------------------------
__global__ __launch_bounds__(128) void k_ffn_out(
    const __bf16* __restrict__ Hb, const float* __restrict__ W_out,
    const float* __restrict__ b_out, const int* __restrict__ poff,
    const int* __restrict__ perm, const float* __restrict__ scl,
    float* __restrict__ out) {
  const int row0 = blockIdx.x * RPB;
  if (row0 >= poff[NE]) return;
  int e = 0;
  while (poff[e + 1] <= row0) ++e;

  const int wave = threadIdx.x >> 5;
  const int lane = threadIdx.x & 31;
  const int n    = lane & 15;
  const int kh   = lane >> 4;
  const int d    = blockIdx.y * 64 + wave * 16 + n;

  const float*  __restrict__ Wo  = W_out + ((size_t)e * DD + d) * DFF;
  const __bf16* __restrict__ Ar0 = Hb + (size_t)(row0 + n) * DFF + kh * 8;
  const __bf16* __restrict__ Ar1 = Ar0 + (size_t)16 * DFF;

  v8f ac0 = {}, ac1 = {};
  for (int k0 = 0; k0 < DFF; k0 += 32) {
    union { uint4 q[2]; v16bf v; } a0, a1;
    a0.q[0] = *(const uint4*)(Ar0 + k0);
    a0.q[1] = *(const uint4*)(Ar0 + k0 + 16);
    a1.q[0] = *(const uint4*)(Ar1 + k0);
    a1.q[1] = *(const uint4*)(Ar1 + k0 + 16);
    const int kb = k0 + kh * 8;
    __builtin_prefetch(Wo + kb + 512, 0, 0);
    const v16bf bw = cvt16(Wo + kb);
    ac0 = __builtin_amdgcn_wmma_f32_16x16x32_bf16(false, a0.v, false, bw,
                                                  (short)0, ac0, false, false);
    ac1 = __builtin_amdgcn_wmma_f32_16x16x32_bf16(false, a1.v, false, bw,
                                                  (short)0, ac1, false, false);
  }

  const float bo = b_out[(size_t)e * DD + d];
#pragma unroll
  for (int r = 0; r < 8; ++r) {
    const int m = r + kh * 8;
    {
      const int srow = row0 + m;
      const int tok  = perm[srow];
      if (tok >= 0)                            // skip pad rows (post-WMMA)
        out[(size_t)tok * DD + d] = scl[srow] * (ac0[r] + bo);
    }
    {
      const int srow = row0 + 16 + m;
      const int tok  = perm[srow];
      if (tok >= 0)
        out[(size_t)tok * DD + d] = scl[srow] * (ac1[r] + bo);
    }
  }
}

// ---------------------------------------------------------------------------
// Launch.  Workspace layout (bytes):
//   [0,256)        poff (9 ints)
//   [256,9472)     perm (2304 ints)
//   [9472,18688)   scl  (2304 floats)
//   [18944,...)    Xb   bf16 [2304 x 1024]  (4.72 MB)
//   [+Xb,...)      Hb   bf16 [2304 x 4096]  (18.9 MB)   total ~23.6 MB
// ---------------------------------------------------------------------------
extern "C" void kernel_launch(void* const* d_in, const int* in_sizes, int n_in,
                              void* d_out, int out_size, void* d_ws, size_t ws_size,
                              hipStream_t stream) {
  const float* hid   = (const float*)d_in[0];
  const float* disp  = (const float*)d_in[1];
  const float* comb  = (const float*)d_in[2];
  const float* W_in  = (const float*)d_in[3];
  const float* b_in  = (const float*)d_in[4];
  const float* W_out = (const float*)d_in[5];
  const float* b_out = (const float*)d_in[6];
  float* out = (float*)d_out;

  char* ws = (char*)d_ws;
  int*    poff = (int*)(ws);
  int*    perm = (int*)(ws + 256);
  float*  scl  = (float*)(ws + 9472);
  __bf16* Xb   = (__bf16*)(ws + 18944);
  __bf16* Hb   = (__bf16*)(ws + 18944 + (size_t)TPAD * DD * 2);

  k_route<<<1, 1024, 0, stream>>>(disp, comb, poff, perm, scl);
  k_gather<<<TPAD / 2, 256, 0, stream>>>(hid, perm, Xb);
  k_ffn_in<<<dim3(ROWT, DFF / 64), 128, 0, stream>>>(Xb, W_in, b_in, poff, Hb);
  k_ffn_out<<<dim3(ROWT, DD / 64), 128, 0, stream>>>(Hb, W_out, b_out, poff,
                                                     perm, scl, out);
}
